// Str2Str_188978561516
// MI455X (gfx1250) — compile-verified, matrix operand order checked by reading, not run on hardware
//
#include <hip/hip_runtime.h>
#include <hip/hip_bf16.h>

#define LSEQ 768
#define TOPK 128
#define DPAIR 128
#define DEDGE 32

typedef __attribute__((ext_vector_type(16))) _Float16 v16h;
typedef __attribute__((ext_vector_type(8)))  _Float16 v8h;
typedef __attribute__((ext_vector_type(8)))  float    v8f;

__device__ inline v8f wmma16(v16h a, v16h b, v8f c) {
  return __builtin_amdgcn_wmma_f32_16x16x32_f16(false, a, false, b, (short)0, c, false, false);
}
// K offset inside a 32-wide chunk for fragment element e, half-wave hi (ISA 16-bit A/B layout)
__device__ inline int frag_k(int e, int hi) { return ((e >> 3) << 4) + (hi << 3) + (e & 7); }

// LDS byte offset of a shared-memory pointer: flat shared address keeps the
// wave-relative LDS offset in addr[31:0] (CDNA5 ISA 10.2 aperture mapping).
__device__ inline unsigned lds_off(void* p) { return (unsigned)(size_t)p; }

// async Global->LDS 16B copy per lane (gfx1250, tracked by ASYNCcnt)
__device__ inline void async_copy_b128(unsigned lds_byte_off, const void* gaddr) {
  asm volatile("global_load_async_to_lds_b128 %0, %1, off"
               :: "v"(lds_byte_off), "v"((unsigned long long)(size_t)gaddr)
               : "memory");
}
__device__ inline void async_wait0() {
  asm volatile("s_wait_asynccnt 0" ::: "memory");
}

// ---------------------------------------------------------------- node features
__global__ __launch_bounds__(64) void node_kernel(
    const float* __restrict__ msa, const float* __restrict__ state,
    const float* __restrict__ w_x, const float* __restrict__ b_x,
    float* __restrict__ node)
{
  const int lane = threadIdx.x & 31;
  const int i = blockIdx.x * 2 + (threadIdx.x >> 5);
  const float* mrow = msa + (size_t)i * 256;   // msa[0,0,i,:]
  float s = 0.f, q = 0.f;
  #pragma unroll
  for (int c = 0; c < 8; ++c) { float x = mrow[lane * 8 + c]; s += x; q += x * x; }
  #pragma unroll
  for (int m = 16; m >= 1; m >>= 1) { s += __shfl_xor(s, m, 32); q += __shfl_xor(q, m, 32); }
  const float mm = s * (1.f / 256.f);
  const float rm = rsqrtf(q * (1.f / 256.f) - mm * mm + 1e-5f);

  const float* srow = state + (size_t)i * 16;
  float xs = (lane < 16) ? srow[lane] : 0.f;
  float ss = xs, qs = xs * xs;
  #pragma unroll
  for (int m = 16; m >= 1; m >>= 1) { ss += __shfl_xor(ss, m, 32); qs += __shfl_xor(qs, m, 32); }
  const float ms_ = ss * (1.f / 16.f);
  const float rs = rsqrtf(qs * (1.f / 16.f) - ms_ * ms_ + 1e-5f);

  float acc = b_x[lane];
  for (int c = 0; c < 256; ++c) acc += (mrow[c] - mm) * rm * w_x[c * 32 + lane];
  #pragma unroll
  for (int c = 0; c < 16; ++c) acc += (srow[c] - ms_) * rs * w_x[(256 + c) * 32 + lane];

  float sn = acc, qn = acc * acc;
  #pragma unroll
  for (int m = 16; m >= 1; m >>= 1) { sn += __shfl_xor(sn, m, 32); qn += __shfl_xor(qn, m, 32); }
  const float mn = sn * (1.f / 32.f);
  const float rn = rsqrtf(qn * (1.f / 32.f) - mn * mn + 1e-5f);
  node[(size_t)i * 32 + lane] = (acc - mn) * rn;
}

// ---------------------------------------------------------------- top-k neighbors
__global__ __launch_bounds__(256) void topk_kernel(
    const float* __restrict__ xyz, const int* __restrict__ idx, int* __restrict__ nbr)
{
  __shared__ unsigned long long keys[1024];
  const int i = blockIdx.x, t = threadIdx.x;
  const float cix = xyz[i * 9 + 3], ciy = xyz[i * 9 + 4], ciz = xyz[i * 9 + 5];
  const int ii = idx[i];
  for (int j = t; j < 1024; j += 256) {
    unsigned long long key;
    if (j < LSEQ) {
      float dx = xyz[j * 9 + 3] - cix, dy = xyz[j * 9 + 4] - ciy, dz = xyz[j * 9 + 5] - ciz;
      float D = sqrtf(dx * dx + dy * dy + dz * dz + 1e-12f);
      int sep = idx[j] - ii; sep = sep < 0 ? -sep : sep;
      float val;
      if (j == i)       val = D + 999.9f;   // eye term
      else if (sep < 9) val = 0.f;          // sequence-local always kept
      else              val = D;
      key = ((unsigned long long)__float_as_uint(val) << 32) | (unsigned)j;
    } else key = ~0ull;
    keys[j] = key;
  }
  __syncthreads();
  for (int k = 2; k <= 1024; k <<= 1)
    for (int jj = k >> 1; jj > 0; jj >>= 1) {
      for (int n = t; n < 1024; n += 256) {
        int ixj = n ^ jj;
        if (ixj > n) {
          bool asc = ((n & k) == 0);
          unsigned long long a = keys[n], b = keys[ixj];
          if ((a > b) == asc) { keys[n] = b; keys[ixj] = a; }
        }
      }
      __syncthreads();
    }
  if (t < TOPK) nbr[i * TOPK + t] = (int)(keys[t] & 0xFFFFFFFFu);
}

// ---------------------------------------------------------------- fused edge pipeline (WMMA)
struct alignas(16) WaveScr {
  float p[16][132];          // gathered pair rows (row stride 528B = 33*16B)
  float mid[16][34];         // GEMM outputs pre-LN
  _Float16 cat[16][96];      // LN(pair_e) | rbf | neigh | zero-pad, f16 (rows 192B)
  float rstat[16][2];
};

__global__ __launch_bounds__(64) void edge_kernel(
    const float* __restrict__ pair, const float* __restrict__ xyz,
    const int* __restrict__ idx, const int* __restrict__ nbr,
    const float* __restrict__ w_e1, const float* __restrict__ b_e1,
    const float* __restrict__ w_e2, const float* __restrict__ b_e2,
    float* __restrict__ edge_out)
{
  __shared__ WaveScr scr[2];
  // fragment-major weights: [kchunk][nchunk][lane][e] -> a lane's whole v16h B
  // fragment is 32 contiguous, 32B-aligned bytes (2x ds_load_b128)
  __shared__ alignas(32) _Float16 we1f[4][2][32][16];
  __shared__ alignas(32) _Float16 we2f[3][2][32][16];
  __shared__ float be1s[32], be2s[32];

  const int tid = threadIdx.x;
  const int w = tid >> 5, lane = tid & 31;
  const int rc = lane & 15;          // row for A, column for B/C
  const int hi = lane >> 4;
  WaveScr& S = scr[w];
  const int tile = blockIdx.x * 2 + w;
  const int i = tile >> 3;
  const int k0 = (tile & 7) << 4;

  // kick off async DMA of the 16 gathered pair rows (cache -> LDS, no VGPR staging)
  for (int r = 0; r < 16; ++r) {
    int j = nbr[i * TOPK + k0 + r];
    const float* src = pair + ((size_t)i * LSEQ + j) * DPAIR + lane * 4;
    async_copy_b128(lds_off(&S.p[r][lane * 4]), src);
  }

  // overlap: swizzle weights to f16 fragment order in LDS while the DMA flies
  for (int t = tid; t < 4 * 2 * 32 * 16; t += 64) {
    int e = t & 15, ln = (t >> 4) & 31, nb = (t >> 9) & 1, kc = t >> 10;
    int K = kc * 32 + frag_k(e, ln >> 4);
    int col = (ln & 15) + nb * 16;
    we1f[kc][nb][ln][e] = (_Float16)w_e1[K * 32 + col];
  }
  for (int t = tid; t < 3 * 2 * 32 * 16; t += 64) {
    int e = t & 15, ln = (t >> 4) & 31, nb = (t >> 9) & 1, kc = t >> 10;
    int K = kc * 32 + frag_k(e, ln >> 4);
    int col = (ln & 15) + nb * 16;
    we2f[kc][nb][ln][e] = (K < 69) ? (_Float16)w_e2[K * 32 + col] : (_Float16)0.f;
  }
  if (tid < 32) { be1s[tid] = b_e1[tid]; be2s[tid] = b_e2[tid]; }

  async_wait0();
  __syncthreads();

  // per-row LN stats over 128 (two half-wave owners per row)
  float s = 0.f, q = 0.f;
  for (int c = hi * 64; c < hi * 64 + 64; ++c) { float x = S.p[rc][c]; s += x; q += x * x; }
  s += __shfl_xor(s, 16, 32); q += __shfl_xor(q, 16, 32);
  const float mean = s * (1.f / 128.f);
  const float rinv = rsqrtf(q * (1.f / 128.f) - mean * mean + 1e-5f);

  // GEMM1: LN(pair) @ w_e1  (16x128x32 via 4 K-chunks x 2 N-chunks)
  v8f acc0 = {}, acc1 = {};
  #pragma unroll
  for (int kc = 0; kc < 4; ++kc) {
    const int kb = kc * 32;
    const float4 p0 = *(const float4*)&S.p[rc][kb + hi * 8];
    const float4 p1 = *(const float4*)&S.p[rc][kb + hi * 8 + 4];
    const float4 p2 = *(const float4*)&S.p[rc][kb + 16 + hi * 8];
    const float4 p3 = *(const float4*)&S.p[rc][kb + 16 + hi * 8 + 4];
    v16h a;
    a[0]  = (_Float16)((p0.x - mean) * rinv); a[1]  = (_Float16)((p0.y - mean) * rinv);
    a[2]  = (_Float16)((p0.z - mean) * rinv); a[3]  = (_Float16)((p0.w - mean) * rinv);
    a[4]  = (_Float16)((p1.x - mean) * rinv); a[5]  = (_Float16)((p1.y - mean) * rinv);
    a[6]  = (_Float16)((p1.z - mean) * rinv); a[7]  = (_Float16)((p1.w - mean) * rinv);
    a[8]  = (_Float16)((p2.x - mean) * rinv); a[9]  = (_Float16)((p2.y - mean) * rinv);
    a[10] = (_Float16)((p2.z - mean) * rinv); a[11] = (_Float16)((p2.w - mean) * rinv);
    a[12] = (_Float16)((p3.x - mean) * rinv); a[13] = (_Float16)((p3.y - mean) * rinv);
    a[14] = (_Float16)((p3.z - mean) * rinv); a[15] = (_Float16)((p3.w - mean) * rinv);
    v16h b0 = *(const v16h*)we1f[kc][0][lane];
    v16h b1 = *(const v16h*)we1f[kc][1][lane];
    acc0 = wmma16(a, b0, acc0);
    acc1 = wmma16(a, b1, acc1);
  }
  #pragma unroll
  for (int r = 0; r < 8; ++r) {
    S.mid[hi * 8 + r][rc]      = acc0[r] + be1s[rc];
    S.mid[hi * 8 + r][rc + 16] = acc1[r] + be1s[rc + 16];
  }
  __syncthreads();

  // LN over 32, write into cat as f16
  float s2 = 0.f, q2 = 0.f;
  #pragma unroll
  for (int c = hi * 16; c < hi * 16 + 16; ++c) { float x = S.mid[rc][c]; s2 += x; q2 += x * x; }
  s2 += __shfl_xor(s2, 16, 32); q2 += __shfl_xor(q2, 16, 32);
  const float mean2 = s2 * (1.f / 32.f);
  const float rinv2 = rsqrtf(q2 * (1.f / 32.f) - mean2 * mean2 + 1e-5f);
  #pragma unroll
  for (int c = hi * 16; c < hi * 16 + 16; ++c)
    S.cat[rc][c] = (_Float16)((S.mid[rc][c] - mean2) * rinv2);

  // RBF(36) + neigh(1) + zero pad to K=96
  {
    int j = nbr[i * TOPK + k0 + rc];
    float dx = xyz[j * 9 + 3] - xyz[i * 9 + 3];
    float dy = xyz[j * 9 + 4] - xyz[i * 9 + 4];
    float dz = xyz[j * 9 + 5] - xyz[i * 9 + 5];
    float D = sqrtf(dx * dx + dy * dy + dz * dz + 1e-12f);
    const float inv_sigma = 36.f / 20.f;
    for (int t = hi * 18; t < hi * 18 + 18; ++t) {
      float mu = 2.f + (20.f / 35.f) * (float)t;   // linspace(2,22,36)
      float z = (D - mu) * inv_sigma;
      S.cat[rc][32 + t] = (_Float16)__expf(-z * z);
    }
    if (hi == 0) {
      int sep = idx[j] - idx[i];
      float ng = (sep >= -1 && sep <= 1) ? (float)sep : 0.f;
      S.cat[rc][68] = (_Float16)ng;
      for (int c = 69; c < 82; ++c) S.cat[rc][c] = (_Float16)0.f;
    } else {
      for (int c = 82; c < 96; ++c) S.cat[rc][c] = (_Float16)0.f;
    }
  }
  __syncthreads();

  // GEMM2: cat @ w_e2  (16x96x32)
  v8f g0 = {}, g1 = {};
  #pragma unroll
  for (int kc = 0; kc < 3; ++kc) {
    const int kb = kc * 32;
    v8h alo = *(const v8h*)&S.cat[rc][kb + hi * 8];       // 16B-aligned
    v8h ahi = *(const v8h*)&S.cat[rc][kb + 16 + hi * 8];  // 16B-aligned
    v16h a;
    #pragma unroll
    for (int e = 0; e < 8; ++e) { a[e] = alo[e]; a[8 + e] = ahi[e]; }
    v16h b0 = *(const v16h*)we2f[kc][0][lane];
    v16h b1 = *(const v16h*)we2f[kc][1][lane];
    g0 = wmma16(a, b0, g0);
    g1 = wmma16(a, b1, g1);
  }
  #pragma unroll
  for (int r = 0; r < 8; ++r) {
    S.mid[hi * 8 + r][rc]      = g0[r] + be2s[rc];
    S.mid[hi * 8 + r][rc + 16] = g1[r] + be2s[rc + 16];
  }
  __syncthreads();

  float s3 = 0.f, q3 = 0.f;
  #pragma unroll
  for (int c = hi * 16; c < hi * 16 + 16; ++c) { float x = S.mid[rc][c]; s3 += x; q3 += x * x; }
  s3 += __shfl_xor(s3, 16, 32); q3 += __shfl_xor(q3, 16, 32);
  if (hi == 0) {
    float m3 = s3 * (1.f / 32.f);
    S.rstat[rc][0] = m3;
    S.rstat[rc][1] = rsqrtf(q3 * (1.f / 32.f) - m3 * m3 + 1e-5f);
  }
  __syncthreads();
  for (int r = 0; r < 16; ++r) {
    float m = S.rstat[r][0], ri = S.rstat[r][1];
    edge_out[((size_t)i * TOPK + k0 + r) * 32 + lane] = (S.mid[r][lane] - m) * ri;
  }
}

// ---------------------------------------------------------------- gated messages (WMMA)
__global__ __launch_bounds__(128) void msg_kernel(
    const float* __restrict__ edge, const float* __restrict__ node,
    const int* __restrict__ nbr,
    const float* __restrict__ Wg, const float* __restrict__ Wm,
    float* __restrict__ m0)
{
  // fragment-major gate/message weights, shared by all 4 waves of the block
  __shared__ alignas(32) _Float16 wgf[32][16];
  __shared__ alignas(32) _Float16 wmf[32][16];
  const int tid = threadIdx.x, lane = tid & 31, w = tid >> 5;
  const int rc = lane & 15, hi = lane >> 4;
  for (int t = tid; t < 512; t += 128) {
    int e = t & 15, ln = t >> 4;
    int K = frag_k(e, ln >> 4), col = ln & 15;
    wgf[ln][e] = (_Float16)Wg[K * 16 + col];
    wmf[ln][e] = (_Float16)Wm[K * 16 + col];
  }
  __syncthreads();

  const int tile = blockIdx.x * 4 + w;
  const int i = tile >> 3, k0 = (tile & 7) << 4;
  const int erow = i * TOPK + k0 + rc;
  const int j = nbr[erow];

  // A fragments: two contiguous 8-float runs each -> global_load_b128 x4
  const float4* er = (const float4*)(edge + (size_t)erow * 32);
  const float4* nr = (const float4*)(node + (size_t)j * 32);
  const float4 e0 = er[hi * 2], e1 = er[hi * 2 + 1], e2 = er[4 + hi * 2], e3 = er[4 + hi * 2 + 1];
  const float4 n0 = nr[hi * 2], n1 = nr[hi * 2 + 1], n2 = nr[4 + hi * 2], n3 = nr[4 + hi * 2 + 1];
  v16h a1, a2;
  a1[0]  = (_Float16)e0.x; a1[1]  = (_Float16)e0.y; a1[2]  = (_Float16)e0.z; a1[3]  = (_Float16)e0.w;
  a1[4]  = (_Float16)e1.x; a1[5]  = (_Float16)e1.y; a1[6]  = (_Float16)e1.z; a1[7]  = (_Float16)e1.w;
  a1[8]  = (_Float16)e2.x; a1[9]  = (_Float16)e2.y; a1[10] = (_Float16)e2.z; a1[11] = (_Float16)e2.w;
  a1[12] = (_Float16)e3.x; a1[13] = (_Float16)e3.y; a1[14] = (_Float16)e3.z; a1[15] = (_Float16)e3.w;
  a2[0]  = (_Float16)n0.x; a2[1]  = (_Float16)n0.y; a2[2]  = (_Float16)n0.z; a2[3]  = (_Float16)n0.w;
  a2[4]  = (_Float16)n1.x; a2[5]  = (_Float16)n1.y; a2[6]  = (_Float16)n1.z; a2[7]  = (_Float16)n1.w;
  a2[8]  = (_Float16)n2.x; a2[9]  = (_Float16)n2.y; a2[10] = (_Float16)n2.z; a2[11] = (_Float16)n2.w;
  a2[12] = (_Float16)n3.x; a2[13] = (_Float16)n3.y; a2[14] = (_Float16)n3.z; a2[15] = (_Float16)n3.w;

  v16h bg = *(const v16h*)wgf[lane];
  v16h bm = *(const v16h*)wmf[lane];
  v8f z0 = {}, z1 = {};
  v8f gl = wmma16(a1, bg, z0);
  v8f ms = wmma16(a2, bm, z1);
  float ssum = 0.f;
  #pragma unroll
  for (int r = 0; r < 8; ++r) {
    float g = 1.f / (1.f + __expf(-gl[r]));
    ssum += ms[r] * g;
  }
  atomicAdd(&m0[i * 16 + rc], ssum);
}

// ---------------------------------------------------------------- h0_out finalize
__global__ void h0_kernel(const float* __restrict__ node, const float* __restrict__ W_self0,
                          const float* __restrict__ m0, float* __restrict__ out)
{
  int t = blockIdx.x * 256 + threadIdx.x;
  if (t >= LSEQ * 16) return;
  int i = t >> 4, n = t & 15;
  float acc = m0[t] * (1.f / (float)TOPK);
  #pragma unroll
  for (int c = 0; c < 32; ++c) acc += node[i * 32 + c] * W_self0[c * 16 + n];
  out[t] = acc;
}

// ---------------------------------------------------------------- SE(3) update
__global__ __launch_bounds__(128) void se3_kernel(
    const float* __restrict__ edge, const int* __restrict__ nbr,
    const float* __restrict__ xyz,
    const float* __restrict__ W_mix1, const float* __restrict__ W_d1,
    const float* __restrict__ W_self1, float* __restrict__ xyz_out)
{
  __shared__ float red[128][6];
  const int i = blockIdx.x, k = threadIdx.x;
  const int j = nbr[i * TOPK + k];

  float wm[6] = {0, 0, 0, 0, 0, 0}, dw[2] = {0, 0};
  for (int c = 0; c < 32; ++c) {
    float x = edge[((size_t)i * TOPK + k) * 32 + c];
    #pragma unroll
    for (int o = 0; o < 6; ++o) wm[o] += x * W_mix1[c * 6 + o];
    dw[0] += x * W_d1[c * 2 + 0];
    dw[1] += x * W_d1[c * 2 + 1];
  }
  float cai[3], caj[3], dvec[3], vj[3][3];
  #pragma unroll
  for (int d = 0; d < 3; ++d) {
    cai[d] = xyz[i * 9 + 3 + d]; caj[d] = xyz[j * 9 + 3 + d];
    dvec[d] = caj[d] - cai[d];
  }
  #pragma unroll
  for (int c = 0; c < 3; ++c)
    #pragma unroll
    for (int d = 0; d < 3; ++d) vj[c][d] = xyz[j * 9 + c * 3 + d] - caj[d];
  #pragma unroll
  for (int o = 0; o < 2; ++o)
    #pragma unroll
    for (int d = 0; d < 3; ++d) {
      float acc = dw[o] * dvec[d];
      #pragma unroll
      for (int c = 0; c < 3; ++c) acc += wm[o * 3 + c] * vj[c][d];
      red[k][o * 3 + d] = acc;
    }
  __syncthreads();
  for (int sft = 64; sft > 0; sft >>= 1) {
    if (k < sft)
      #pragma unroll
      for (int o = 0; o < 6; ++o) red[k][o] += red[k + sft][o];
    __syncthreads();
  }
  if (k == 0) {
    float vi[3][3];
    #pragma unroll
    for (int c = 0; c < 3; ++c)
      #pragma unroll
      for (int d = 0; d < 3; ++d) vi[c][d] = xyz[i * 9 + c * 3 + d] - cai[d];
    float m1[2][3];
    #pragma unroll
    for (int o = 0; o < 2; ++o)
      #pragma unroll
      for (int d = 0; d < 3; ++d) {
        float acc = red[0][o * 3 + d] * (1.f / (float)TOPK);
        #pragma unroll
        for (int c = 0; c < 3; ++c) acc += W_self1[o * 3 + c] * vi[c][d];
        m1[o][d] = acc * 0.01f;   // offset = m1 / 100
      }
    float R[3] = {m1[1][0], m1[1][1], m1[1][2]};
    float Rang = sqrtf(R[0] * R[0] + R[1] * R[1] + R[2] * R[2]);
    float rn = 1.f / (Rang + 1e-5f);
    float Rv[3] = {R[0] * rn, R[1] * rn, R[2] * rn};
    float cA = cosf(Rang), sA = sinf(Rang);
    #pragma unroll
    for (int c = 0; c < 3; ++c) {
      float dot = Rv[0] * vi[c][0] + Rv[1] * vi[c][1] + Rv[2] * vi[c][2];
      float cr[3] = {Rv[1] * vi[c][2] - Rv[2] * vi[c][1],
                     Rv[2] * vi[c][0] - Rv[0] * vi[c][2],
                     Rv[0] * vi[c][1] - Rv[1] * vi[c][0]};
      #pragma unroll
      for (int d = 0; d < 3; ++d) {
        float vperp = vi[c][d] - Rv[d] * dot;
        float vnew = vperp * cA + cr[d] * sA + Rv[d] * dot;
        xyz_out[i * 9 + c * 3 + d] = vnew + cai[d] + m1[0][d];
      }
    }
  }
}

__global__ void zero_kernel(float* __restrict__ p, int n) {
  int t = blockIdx.x * 256 + threadIdx.x;
  if (t < n) p[t] = 0.f;
}

// ---------------------------------------------------------------- launch
extern "C" void kernel_launch(void* const* d_in, const int* in_sizes, int n_in,
                              void* d_out, int out_size, void* d_ws, size_t ws_size,
                              hipStream_t stream) {
  const float* msa    = (const float*)d_in[0];
  const float* pair   = (const float*)d_in[1];
  const float* xyz    = (const float*)d_in[2];
  const float* state  = (const float*)d_in[3];
  const int*   idx    = (const int*)d_in[4];
  // d_in[5] = top_k (compile-time TOPK)
  const float* w_x    = (const float*)d_in[6];
  const float* b_x    = (const float*)d_in[7];
  const float* w_e1   = (const float*)d_in[8];
  const float* b_e1   = (const float*)d_in[9];
  const float* w_e2   = (const float*)d_in[10];
  const float* b_e2   = (const float*)d_in[11];
  const float* W_self0 = (const float*)d_in[12];
  const float* W_msg0  = (const float*)d_in[13];
  const float* W_gate0 = (const float*)d_in[14];
  const float* W_mix1  = (const float*)d_in[15];
  const float* W_d1    = (const float*)d_in[16];
  const float* W_self1 = (const float*)d_in[17];
  float* out = (float*)d_out;

  char* ws = (char*)d_ws;
  float* node = (float*)ws;                                       // 768*32  f32
  int*   nbr  = (int*)(ws + 98304);                               // 768*128 i32
  float* edge = (float*)(ws + 98304 + 393216);                    // 768*128*32 f32
  float* m0   = (float*)(ws + 98304 + 393216 + 12582912);         // 768*16  f32

  node_kernel<<<LSEQ / 2, 64, 0, stream>>>(msa, state, w_x, b_x, node);
  topk_kernel<<<LSEQ, 256, 0, stream>>>(xyz, idx, nbr);
  zero_kernel<<<48, 256, 0, stream>>>(m0, LSEQ * 16);
  edge_kernel<<<(LSEQ * 8) / 2, 64, 0, stream>>>(pair, xyz, idx, nbr,
                                                 w_e1, b_e1, w_e2, b_e2, edge);
  msg_kernel<<<(LSEQ * 8) / 4, 128, 0, stream>>>(edge, node, nbr, W_gate0, W_msg0, m0);
  h0_kernel<<<48, 256, 0, stream>>>(node, W_self0, m0, out + LSEQ * 9);
  se3_kernel<<<LSEQ, 128, 0, stream>>>(edge, nbr, xyz, W_mix1, W_d1, W_self1, out);
}
